// SuperpixelContrastiveLearning_53704271069797
// MI455X (gfx1250) — compile-verified, matrix operand order checked by reading, not run on hardware
//
#include <hip/hip_runtime.h>
#include <hip/hip_bf16.h>

// ---------------- problem constants (match reference) ----------------
#define NSP   1024
#define FDIM  256
#define HDIM  128
#define ODIM  64
#define BATCH 4
#define IMG_H 256
#define IMG_W 256
#define HWPIX (IMG_H * IMG_W)
#define INV_TEMP 10.0f   // 1 / TEMPERATURE

typedef __attribute__((ext_vector_type(16))) __bf16 v16bf;
typedef __attribute__((ext_vector_type(8)))  __bf16 v8bf;
typedef __attribute__((ext_vector_type(8)))  float  v8f;
typedef __attribute__((ext_vector_type(4)))  unsigned int u32x4;
typedef __attribute__((ext_vector_type(8)))  int   i32x8;
typedef __attribute__((ext_vector_type(4)))  int   i32x4;

#if defined(__gfx1250__) && __has_builtin(__builtin_amdgcn_tensor_load_to_lds)
#define USE_TDM 1
#else
#define USE_TDM 0
#endif

#if USE_TDM
// ---------------------------------------------------------------------
// TDM: DMA a [16 rows x 32 contiguous cols] bf16 tile from a row-major
// [N x K] tensor (row stride ldbt elements) into LDS at lds_addr.
// D# packing per CDNA5 ISA 8.3/8.4 (2D tensor, groups 2/3 zero).
// ---------------------------------------------------------------------
__device__ __forceinline__ void tdm_load_bt_tile(unsigned lds_addr,
                                                 const __bf16* gbase,
                                                 long long elem_off, int ldbt,
                                                 int tensor_d0, int tensor_d1) {
    const unsigned long long ga =
        (unsigned long long)gbase + ((unsigned long long)elem_off << 1);
    u32x4 g0;
    g0[0] = 1u;                                   // count=1, is_restore=0
    g0[1] = lds_addr;                             // LDS byte address
    g0[2] = (unsigned)(ga & 0xffffffffull);       // global_addr[31:0]
    g0[3] = (unsigned)((ga >> 32) & 0x01ffffffull) | 0x80000000u; // addr[56:32], type=2
    i32x8 g1;
    g1[0] = (1 << 16);                            // data_size=1 (2 bytes), no flags
    g1[1] = (int)(((unsigned)tensor_d0 & 0xffffu) << 16);       // tensor_dim0[15:0]
    g1[2] = (int)((((unsigned)tensor_d0 >> 16) & 0xffffu) |
                  (((unsigned)tensor_d1 & 0xffffu) << 16));     // td0[31:16] | td1[15:0]
    g1[3] = (int)((((unsigned)tensor_d1 >> 16) & 0xffffu) |
                  (32u << 16));                                 // td1[31:16] | tile_dim0=32
    g1[4] = 16;                                   // tile_dim1=16, tile_dim2=0
    g1[5] = ldbt;                                 // tensor_dim0_stride[31:0]
    g1[6] = 0;                                    // stride[47:32], dim1_stride lo
    g1[7] = 0;
    const i32x4 z4 = {0, 0, 0, 0};
#if defined(__clang_major__) && __clang_major__ >= 23
    const i32x8 z8 = {0, 0, 0, 0, 0, 0, 0, 0};
    __builtin_amdgcn_tensor_load_to_lds(g0, g1, z4, z4, z8, 0);
#else
    __builtin_amdgcn_tensor_load_to_lds(g0, g1, z4, z4, 0);
#endif
}
#endif // USE_TDM

// pack two contiguous 8-element runs into the WMMA v16bf operand
__device__ __forceinline__ v16bf pack16(v8bf lo, v8bf hi) {
    return __builtin_shufflevector(lo, hi, 0, 1, 2, 3, 4, 5, 6, 7,
                                   8, 9, 10, 11, 12, 13, 14, 15);
}

// =====================================================================
// 1a) Superpixel pixel counts
// =====================================================================
__global__ void count_kernel(const int* __restrict__ spidx,
                             float* __restrict__ counts) {
    const int p = blockIdx.x * blockDim.x + threadIdx.x;
    const int b = blockIdx.y;
    if (p >= HWPIX) return;
    atomicAdd(&counts[b * NSP + spidx[(size_t)b * HWPIX + p]], 1.0f);
}

// =====================================================================
// 1b) Aggregation with LDS accumulation (ds_add_f32)
// =====================================================================
#define AGG_CH  8
#define AGG_PIX 16384
__global__ void agg_lds_kernel(const float* __restrict__ feat,
                               const int* __restrict__ spidx,
                               float* __restrict__ sp_sum) {
    __shared__ float acc[NSP * AGG_CH];                  // 32 KB
    const int b  = blockIdx.z;
    const int c0 = blockIdx.y * AGG_CH;
    const int p0 = blockIdx.x * AGG_PIX;
    const int tid = threadIdx.x;
    for (int i = tid; i < NSP * AGG_CH; i += 256) acc[i] = 0.0f;
    __syncthreads();
    const int*   im = spidx + (size_t)b * HWPIX;
    const float* f  = feat  + (size_t)b * FDIM * HWPIX + (size_t)c0 * HWPIX;
    for (int p = p0 + tid; p < p0 + AGG_PIX; p += 256) {
        const int s = im[p];
        #pragma unroll
        for (int c = 0; c < AGG_CH; ++c)
            atomicAdd(&acc[s * AGG_CH + c], f[(size_t)c * HWPIX + p]);
    }
    __syncthreads();
    for (int i = tid; i < NSP * AGG_CH; i += 256) {
        const int s = i / AGG_CH, c = i % AGG_CH;
        atomicAdd(&sp_sum[((size_t)b * NSP + s) * FDIM + c0 + c], acc[i]);
    }
}

// mean + bf16, written TRANSPOSED: spfT[b][c][s]  (B^T operand for GEMM1)
__global__ void agg_finalize_kernel(const float* __restrict__ sp_sum,
                                    const float* __restrict__ counts,
                                    __bf16* __restrict__ spfT) {
    const size_t i = (size_t)blockIdx.x * blockDim.x + threadIdx.x; // output-linear
    const int s  = (int)(i & (NSP - 1));
    const size_t bc = i >> 10;
    const int c  = (int)(bc & (FDIM - 1));
    const int b  = (int)(bc >> 8);
    const float cnt = counts[b * NSP + s];
    const float v = (cnt > 0.0f) ? sp_sum[((size_t)b * NSP + s) * FDIM + c] / cnt : 0.0f;
    spfT[i] = (__bf16)v;
}

// =====================================================================
// 2) Adjacency
// =====================================================================
__global__ void adj_mark_kernel(const int* __restrict__ spidx,
                                float* __restrict__ adjf) {
    const int p = blockIdx.x * blockDim.x + threadIdx.x;
    const int b = blockIdx.y;
    if (p >= HWPIX) return;
    const int h = p / IMG_W, w = p % IMG_W;
    const int* im = spidx + (size_t)b * HWPIX;
    float* adj = adjf + (size_t)b * NSP * NSP;
    const int s = im[p];
    if (w + 1 < IMG_W) {
        const int s2 = im[p + 1];
        if (s != s2) { adj[(size_t)s * NSP + s2] = 1.0f; adj[(size_t)s2 * NSP + s] = 1.0f; }
    }
    if (h + 1 < IMG_H) {
        const int s2 = im[p + IMG_W];
        if (s != s2) { adj[(size_t)s * NSP + s2] = 1.0f; adj[(size_t)s2 * NSP + s] = 1.0f; }
    }
}

__global__ void adj_degree_kernel(float* __restrict__ adjf,
                                  float* __restrict__ deg) {
    __shared__ float red[256];
    const int row = blockIdx.x, b = blockIdx.y, tid = threadIdx.x;
    float* a = adjf + ((size_t)b * NSP + row) * NSP;
    if (tid == 0) a[row] = 1.0f;                  // + eye
    __syncthreads();
    float s = 0.0f;
    for (int j = tid; j < NSP; j += 256) s += a[j];
    red[tid] = s; __syncthreads();
    for (int off = 128; off > 0; off >>= 1) {
        if (tid < off) red[tid] += red[tid + off];
        __syncthreads();
    }
    if (tid == 0) deg[b * NSP + row] = red[0];
}

// note: adjacency is symmetric, so adjb == adjb^T; it serves as A directly
__global__ void adj_norm_kernel(const float* __restrict__ adjf,
                                const float* __restrict__ deg,
                                __bf16* __restrict__ adjb) {
    const size_t i = (size_t)blockIdx.x * blockDim.x + threadIdx.x;
    const int b = blockIdx.y;
    const int row = (int)(i >> 10);
    const int col = (int)(i & (NSP - 1));
    const float d = rsqrtf(deg[b * NSP + row]) * rsqrtf(deg[b * NSP + col]);
    const size_t o = (size_t)b * NSP * NSP + i;
    adjb[o] = (__bf16)(adjf[o] * d);
}

// =====================================================================
// 3) Weight f32 -> bf16 copy (weights are already [N][K] = B^T layout)
// =====================================================================
__global__ void wcvt_kernel(const float* __restrict__ in, __bf16* __restrict__ out) {
    const int i = blockIdx.x * blockDim.x + threadIdx.x;
    out[i] = (__bf16)in[i];
}

// =====================================================================
// 4) Batched WMMA GEMM (NT):  C[m,n] = act(scale * sum_k A[m,k]*Bt[n,k] + bias)
//    A [M,K] and Bt [N,K] bf16 row-major. Wave per 16x16 tile.
//    Bt tiles TDM-staged into per-wave double-buffered LDS; all operand
//    fetches are 128-bit (2x global b128 + 2x ds b128 per lane per WMMA).
// =====================================================================
template<bool RELU, bool HAS_BIAS, bool OUT_F32, bool OUT_BF16>
__global__ void gemm_wmma_bf16_kernel(const __bf16* __restrict__ A, long long sA, int lda,
                                      const __bf16* __restrict__ Bt, long long sB, int ldbt,
                                      const float* __restrict__ bias,
                                      float* __restrict__ Cf, __bf16* __restrict__ Cbf,
                                      long long sC, int ldc,
                                      int M, int N, int K, float scale) {
#if USE_TDM
    __shared__ alignas(128) __bf16 Btile[4 * 2 * 512]; // 4 waves x 2 bufs x (16x32) bf16
#endif
    const int bimg = blockIdx.y;
    const __bf16* Ab  = A  + (size_t)bimg * sA;
    const __bf16* Btb = Bt + (size_t)bimg * sB;

    const int wave = (blockIdx.x * blockDim.x + threadIdx.x) >> 5;
    const int lane = threadIdx.x & 31;
    const int lh   = lane >> 4;          // lane half
    const int r    = lane & 15;
    const int tilesN = N >> 4;
    const int m0 = (wave / tilesN) << 4;
    const int n0 = (wave % tilesN) << 4;
    const int nk = K >> 5;

    // per-lane A row base: covers k runs [k0+8*lh .. +7] and [k0+16+8*lh .. +7]
    const __bf16* arow = Ab + (size_t)(m0 + r) * lda + (lh << 3);

    v8f acc = {};
#if USE_TDM
    const int wv = __builtin_amdgcn_readfirstlane((int)(threadIdx.x >> 5));
    const unsigned ldsW = (unsigned)(unsigned long long)(&Btile[0]) + (unsigned)wv * 2048u;
    // prime buffer 0 with rows n0..n0+15, cols 0..31
    tdm_load_bt_tile(ldsW, Btb, (long long)n0 * ldbt, ldbt, K, N);
    for (int it = 0; it < nk; ++it) {
        const int cur = it & 1;
        if (it + 1 < nk) {
            tdm_load_bt_tile(ldsW + (unsigned)(cur ^ 1) * 1024u, Btb,
                             (long long)n0 * ldbt + (long long)(it + 1) * 32, ldbt, K, N);
            asm volatile("s_wait_tensorcnt 0x1" ::: "memory");   // cur tile landed
        } else {
            asm volatile("s_wait_tensorcnt 0x0" ::: "memory");
        }
        const int k0 = it << 5;
        // LDS tile row r, two 16B runs at k offsets 8*lh and 16+8*lh
        const __bf16* bl = &Btile[wv * 1024 + cur * 512] + r * 32 + (lh << 3);
        const v8bf a0 = *(const v8bf*)(arow + k0);
        const v8bf a1 = *(const v8bf*)(arow + k0 + 16);
        const v8bf b0 = *(const v8bf*)(bl);
        const v8bf b1 = *(const v8bf*)(bl + 16);
        acc = __builtin_amdgcn_wmma_f32_16x16x32_bf16(
                  false, pack16(a0, a1), false, pack16(b0, b1),
                  (short)0, acc, false, false);
    }
#else
    const __bf16* brow = Btb + (size_t)(n0 + r) * ldbt + (lh << 3);
    for (int it = 0; it < nk; ++it) {
        const int k0 = it << 5;
        const v8bf a0 = *(const v8bf*)(arow + k0);
        const v8bf a1 = *(const v8bf*)(arow + k0 + 16);
        const v8bf b0 = *(const v8bf*)(brow + k0);
        const v8bf b1 = *(const v8bf*)(brow + k0 + 16);
        acc = __builtin_amdgcn_wmma_f32_16x16x32_bf16(
                  false, pack16(a0, a1), false, pack16(b0, b1),
                  (short)0, acc, false, false);
    }
#endif

    const int n = n0 + r;
    float bv = 0.0f;
    if (HAS_BIAS) bv = bias[n];
    #pragma unroll
    for (int v = 0; v < 8; ++v) {
        const int m = m0 + v + (lh << 3);   // C/D stripe: VGPR v -> M = v + 8*laneHalf
        float val = acc[v] * scale;
        if (HAS_BIAS) val += bv;
        if (RELU) val = fmaxf(val, 0.0f);
        if (OUT_F32)  Cf [(size_t)bimg * sC + (size_t)m * ldc + n] = val;
        if (OUT_BF16) Cbf[(size_t)bimg * sC + (size_t)m * ldc + n] = (__bf16)val;
    }
}

// =====================================================================
// 5) Row L2 normalize z1, z2 -> z1n, z2n (bf16, both row-major [NSP][ODIM]);
//    z2n's natural layout IS the sim GEMM's B^T. Wave per row.
// =====================================================================
__global__ void znorm_kernel(const float* __restrict__ z1, const float* __restrict__ z2,
                             __bf16* __restrict__ z1n, __bf16* __restrict__ z2n) {
    const int wid  = (blockIdx.x * blockDim.x + threadIdx.x) >> 5;
    const int lane = threadIdx.x & 31;
    const float* p1 = z1 + (size_t)wid * ODIM;
    const float* p2 = z2 + (size_t)wid * ODIM;
    const float a0 = p1[lane], a1 = p1[lane + 32];
    const float c0 = p2[lane], c1 = p2[lane + 32];
    float s1 = a0 * a0 + a1 * a1;
    float s2 = c0 * c0 + c1 * c1;
    #pragma unroll
    for (int m = 16; m > 0; m >>= 1) {
        s1 += __shfl_xor(s1, m, 32);
        s2 += __shfl_xor(s2, m, 32);
    }
    const float i1 = rsqrtf(s1), i2 = rsqrtf(s2);
    z1n[(size_t)wid * ODIM + lane]      = (__bf16)(a0 * i1);
    z1n[(size_t)wid * ODIM + lane + 32] = (__bf16)(a1 * i1);
    z2n[(size_t)wid * ODIM + lane]      = (__bf16)(c0 * i2);
    z2n[(size_t)wid * ODIM + lane + 32] = (__bf16)(c1 * i2);
}

// =====================================================================
// 6) Per-row InfoNCE: loss_row = logsumexp(sim_row) - sim[row][row]
// =====================================================================
__global__ void row_loss_kernel(const float* __restrict__ sim, float* __restrict__ out) {
    __shared__ float red[256];
    const int row = blockIdx.x, b = blockIdx.y, tid = threadIdx.x;
    const float* s = sim + ((size_t)b * NSP + row) * NSP;
    float mx = -3.402823e38f;
    for (int j = tid; j < NSP; j += 256) mx = fmaxf(mx, s[j]);
    red[tid] = mx; __syncthreads();
    for (int off = 128; off > 0; off >>= 1) {
        if (tid < off) red[tid] = fmaxf(red[tid], red[tid + off]);
        __syncthreads();
    }
    mx = red[0]; __syncthreads();
    float sum = 0.0f;
    for (int j = tid; j < NSP; j += 256) sum += __expf(s[j] - mx);
    red[tid] = sum; __syncthreads();
    for (int off = 128; off > 0; off >>= 1) {
        if (tid < off) red[tid] += red[tid + off];
        __syncthreads();
    }
    if (tid == 0) {
        const float lse = mx + __logf(red[0]);
        atomicAdd(out, (lse - s[row]) * (1.0f / (BATCH * NSP)));
    }
}

// =====================================================================
// host-side launch
// =====================================================================
extern "C" void kernel_launch(void* const* d_in, const int* in_sizes, int n_in,
                              void* d_out, int out_size, void* d_ws, size_t ws_size,
                              hipStream_t stream) {
    const float* features = (const float*)d_in[0];
    const int*   spidx    = (const int*)d_in[1];
    const float* W1  = (const float*)d_in[2];
    const float* b1  = (const float*)d_in[3];
    const float* W2  = (const float*)d_in[4];
    const float* b2  = (const float*)d_in[5];
    const float* Wp1 = (const float*)d_in[6];
    const float* bp1 = (const float*)d_in[7];
    const float* Wp2 = (const float*)d_in[8];
    const float* bp2 = (const float*)d_in[9];
    float* out = (float*)d_out;

    size_t off = 0;
    auto carve = [&](size_t bytes) -> void* {
        void* p = (char*)d_ws + off;
        off += (bytes + 255) & ~(size_t)255;
        return p;
    };
    float*  sp_sum = (float*) carve((size_t)BATCH * NSP * FDIM * 4);
    float*  counts = (float*) carve((size_t)BATCH * NSP * 4);
    float*  adjf   = (float*) carve((size_t)BATCH * NSP * NSP * 4);   // reused as sim
    float*  deg    = (float*) carve((size_t)BATCH * NSP * 4);
    __bf16* adjb   = (__bf16*)carve((size_t)BATCH * NSP * NSP * 2);
    __bf16* spfT   = (__bf16*)carve((size_t)BATCH * FDIM * NSP * 2);  // [b][c][s] = B^T
    __bf16* xb     = (__bf16*)carve((size_t)BATCH * NSP * FDIM * 2);
    __bf16* h1b    = (__bf16*)carve((size_t)BATCH * NSP * HDIM * 2);
    __bf16* h2b    = (__bf16*)carve((size_t)BATCH * NSP * HDIM * 2);
    float*  z1     = (float*) carve((size_t)BATCH * NSP * ODIM * 4);
    float*  z2     = (float*) carve((size_t)BATCH * NSP * ODIM * 4);
    __bf16* z1nb   = (__bf16*)carve((size_t)BATCH * NSP * ODIM * 2);
    __bf16* z2nb   = (__bf16*)carve((size_t)BATCH * NSP * ODIM * 2);
    __bf16* w1b    = (__bf16*)carve((size_t)HDIM * FDIM * 2);         // [N][K] natural
    __bf16* w2b    = (__bf16*)carve((size_t)HDIM * HDIM * 2);
    __bf16* wp1b   = (__bf16*)carve((size_t)ODIM * HDIM * 2);
    __bf16* wp2b   = (__bf16*)carve((size_t)ODIM * HDIM * 2);
    float*  simf   = adjf;

    hipMemsetAsync(sp_sum, 0, (size_t)BATCH * NSP * FDIM * 4, stream);
    hipMemsetAsync(counts, 0, (size_t)BATCH * NSP * 4, stream);
    hipMemsetAsync(adjf,   0, (size_t)BATCH * NSP * NSP * 4, stream);
    hipMemsetAsync(out,    0, sizeof(float), stream);

    // ---- stage 1: aggregation (LDS-accumulated) ----
    count_kernel<<<dim3(HWPIX / 256, BATCH), 256, 0, stream>>>(spidx, counts);
    agg_lds_kernel<<<dim3(HWPIX / AGG_PIX, FDIM / AGG_CH, BATCH), 256, 0, stream>>>(
        features, spidx, sp_sum);
    agg_finalize_kernel<<<(BATCH * NSP * FDIM) / 256, 256, 0, stream>>>(sp_sum, counts, spfT);

    // ---- stage 2: adjacency ----
    adj_mark_kernel<<<dim3(HWPIX / 256, BATCH), 256, 0, stream>>>(spidx, adjf);
    adj_degree_kernel<<<dim3(NSP, BATCH), 256, 0, stream>>>(adjf, deg);
    adj_norm_kernel<<<dim3((NSP * NSP) / 256, BATCH), 256, 0, stream>>>(adjf, deg, adjb);

    // ---- stage 3: weight bf16 conversion (already B^T layout) ----
    wcvt_kernel<<<(HDIM * FDIM) / 256, 256, 0, stream>>>(W1,  w1b);
    wcvt_kernel<<<(HDIM * HDIM) / 256, 256, 0, stream>>>(W2,  w2b);
    wcvt_kernel<<<(ODIM * HDIM) / 256, 256, 0, stream>>>(Wp1, wp1b);
    wcvt_kernel<<<(ODIM * HDIM) / 256, 256, 0, stream>>>(Wp2, wp2b);

    // ---- stage 4: GEMM chain (NT: second operand is B^T [N][K]) ----
    // x = adj @ sp_feat : Bt = spfT [FDIM][NSP]
    gemm_wmma_bf16_kernel<false,false,false,true><<<dim3((64*16)/4, BATCH), 128, 0, stream>>>(
        adjb, (long long)NSP*NSP, NSP, spfT, (long long)FDIM*NSP, NSP,
        nullptr, nullptr, xb, (long long)NSP*FDIM, FDIM, NSP, FDIM, NSP, 1.0f);
    // h1 = relu(x @ W1^T + b1) : Bt = W1 [HDIM][FDIM]
    gemm_wmma_bf16_kernel<true,true,false,true><<<dim3((64*8)/4, BATCH), 128, 0, stream>>>(
        xb, (long long)NSP*FDIM, FDIM, w1b, 0LL, FDIM,
        b1, nullptr, h1b, (long long)NSP*HDIM, HDIM, NSP, HDIM, FDIM, 1.0f);
    // h2 = relu(h1 @ W2^T + b2) : Bt = W2 [HDIM][HDIM]
    gemm_wmma_bf16_kernel<true,true,false,true><<<dim3((64*8)/4, BATCH), 128, 0, stream>>>(
        h1b, (long long)NSP*HDIM, HDIM, w2b, 0LL, HDIM,
        b2, nullptr, h2b, (long long)NSP*HDIM, HDIM, NSP, HDIM, HDIM, 1.0f);
    // z1 = h2 @ Wp1^T + bp1 : Bt = Wp1 [ODIM][HDIM]
    gemm_wmma_bf16_kernel<false,true,true,false><<<dim3((64*4)/4, BATCH), 128, 0, stream>>>(
        h2b, (long long)NSP*HDIM, HDIM, wp1b, 0LL, HDIM,
        bp1, z1, nullptr, (long long)NSP*ODIM, ODIM, NSP, ODIM, HDIM, 1.0f);
    // z2 = h2 @ Wp2^T + bp2
    gemm_wmma_bf16_kernel<false,true,true,false><<<dim3((64*4)/4, BATCH), 128, 0, stream>>>(
        h2b, (long long)NSP*HDIM, HDIM, wp2b, 0LL, HDIM,
        bp2, z2, nullptr, (long long)NSP*ODIM, ODIM, NSP, ODIM, HDIM, 1.0f);

    // ---- stage 5: normalize + similarity ----
    znorm_kernel<<<(BATCH * NSP) / 4, 128, 0, stream>>>(z1, z2, z1nb, z2nb);
    // sim = (z1n @ z2n^T) / T : Bt = z2n [NSP][ODIM] (natural layout)
    gemm_wmma_bf16_kernel<false,false,true,false><<<dim3((64*64)/4, BATCH), 128, 0, stream>>>(
        z1nb, (long long)NSP*ODIM, ODIM, z2nb, (long long)NSP*ODIM, ODIM,
        nullptr, simf, nullptr, (long long)NSP*NSP, NSP, NSP, NSP, ODIM, INV_TEMP);

    // ---- stage 6: InfoNCE ----
    row_loss_kernel<<<dim3(NSP, BATCH), 256, 0, stream>>>(simf, out);
}
